// HybridGNN_34548716929465
// MI455X (gfx1250) — matrix operand order repaired and checked by reading, not compiled, not run on hardware
//
#include <hip/hip_runtime.h>

#define HIDN 128
#define NUSER 50000
#define NMERCH 50000
#define DIN 64

typedef __attribute__((ext_vector_type(16))) __bf16 v16bf;
typedef __attribute__((ext_vector_type(8)))  float  v8f;
typedef __attribute__((ext_vector_type(4)))  unsigned int u32x4;

union FragBF { u32x4 q[2]; v16bf v; };

__device__ __forceinline__ unsigned short f2bf(float f) {
  unsigned u = __float_as_uint(f);
  u += 0x7FFFu + ((u >> 16) & 1u);          // round-to-nearest-even
  return (unsigned short)(u >> 16);
}
__device__ __forceinline__ float bf2f(unsigned short h) {
  return __uint_as_float(((unsigned)h) << 16);
}

// ---------------- utility kernels ----------------

__global__ void k_zero_f32(float* __restrict__ p, long long n) {
  long long i = (long long)blockIdx.x * blockDim.x + threadIdx.x;
  long long stride = (long long)gridDim.x * blockDim.x;
  for (; i < n; i += stride) p[i] = 0.f;
}

__global__ void k_cvt_bf16(const float* __restrict__ s, unsigned short* __restrict__ d, int n) {
  int i = blockIdx.x * blockDim.x + threadIdx.x;
  if (i < n) d[i] = f2bf(s[i]);
}

__global__ void k_count(const int* __restrict__ ei, int E, float* __restrict__ cnt, int nDst) {
  int e = blockIdx.x * blockDim.x + threadIdx.x;
  if (e >= E) return;
  int c = ei[E + e];
  if ((unsigned)c < (unsigned)nDst) atomicAdd(&cnt[c], 1.f);
}

// scatter-add of src[row[e]] into agg[col[e]], one thread per (edge, feature)
template<int D, bool SRCF32>
__global__ void k_scatter(const int* __restrict__ ei, int E, const void* __restrict__ src,
                          float* __restrict__ agg, int nSrc, int nDst) {
  long long idx = (long long)blockIdx.x * blockDim.x + threadIdx.x;
  if (idx >= (long long)E * D) return;
  int e = (int)(idx / D);
  int k = (int)(idx & (D - 1));
  int r = ei[e], c = ei[E + e];
  if ((unsigned)r >= (unsigned)nSrc || (unsigned)c >= (unsigned)nDst) return;
  float v = SRCF32 ? ((const float*)src)[(long long)r * D + k]
                   : bf2f(((const unsigned short*)src)[(long long)r * D + k]);
  atomicAdd(&agg[(long long)c * D + k], v);
}

// ---------------- fused SAGE layer: out = act( [mean|xdst] @ [Wl|Wr]^T + b ) ----------------
// Block: 128 threads (4 waves), computes 16 rows x 128 cols. K = 2*D.
template<int D, bool XF32, bool RELU>
__global__ __launch_bounds__(128)
void k_sage_gemm(const float* __restrict__ agg, const float* __restrict__ cnt,
                 const void* __restrict__ xdst_, const unsigned short* __restrict__ wl,
                 const unsigned short* __restrict__ wr, const float* __restrict__ bias,
                 unsigned short* __restrict__ out, int nDst) {
  constexpr int K = 2 * D;
  __shared__ __align__(16) unsigned short At[16 * K];
  const int tid = threadIdx.x;
  const int rowBase = blockIdx.x * 16;

  // build bf16 A-tile (mean | x_dst) in LDS
  for (int i = tid; i < 16 * K; i += 128) {
    int r = i / K, k = i - r * K;
    int gr = rowBase + r;
    if (gr >= nDst) gr = nDst - 1;
    float v;
    if (k < D) {
      float c = cnt[gr];
      float m = c > 1.f ? c : 1.f;
      v = agg[(long long)gr * D + k] / m;
    } else {
      if (XF32) v = ((const float*)xdst_)[(long long)gr * D + (k - D)];
      else      v = bf2f(((const unsigned short*)xdst_)[(long long)gr * D + (k - D)]);
    }
    At[i] = f2bf(v);
  }
  __syncthreads();

  const int lane = tid & 31, wave = tid >> 5;
  const int lrow = lane & 15, hi = lane >> 4;
  v8f acc0 = {0,0,0,0,0,0,0,0};
  v8f acc1 = {0,0,0,0,0,0,0,0};

  for (int k0 = 0; k0 < K; k0 += 32) {
    // A frag: 16x32 bf16 layout (lane<16: K 0..15 ; lane>=16: K 16..31)
    FragBF a;
    const unsigned short* ar = &At[lrow * K + k0 + hi * 16];
    a.q[0] = *(const u32x4*)ar;
    a.q[1] = *(const u32x4*)(ar + 8);

    const unsigned short* wsel = (k0 < D) ? wl : wr;
    const int kk = (k0 < D) ? k0 : (k0 - D);
    {
      int n = (wave * 2 + 0) * 16 + lrow;
      const unsigned short* br = &wsel[(long long)n * D + kk + hi * 16];
      FragBF b; b.q[0] = *(const u32x4*)br; b.q[1] = *(const u32x4*)(br + 8);
      acc0 = __builtin_amdgcn_wmma_f32_16x16x32_bf16(false, a.v, false, b.v,
                                                     (short)0, acc0, false, false);
    }
    {
      int n = (wave * 2 + 1) * 16 + lrow;
      const unsigned short* br = &wsel[(long long)n * D + kk + hi * 16];
      FragBF b; b.q[0] = *(const u32x4*)br; b.q[1] = *(const u32x4*)(br + 8);
      acc1 = __builtin_amdgcn_wmma_f32_16x16x32_bf16(false, a.v, false, b.v,
                                                     (short)0, acc1, false, false);
    }
  }

  // C/D layout: lane<16 -> N=lane, vgpr v -> M=v ; lane>=16 -> N=lane-16, M=v+8
  #pragma unroll
  for (int t = 0; t < 2; ++t) {
    v8f acc = t ? acc1 : acc0;
    int col = (wave * 2 + t) * 16 + lrow;
    float bv = bias[col];
    #pragma unroll
    for (int v = 0; v < 8; ++v) {
      int r = hi ? (v + 8) : v;
      int gr = rowBase + r;
      if (gr < nDst) {
        float x = acc[v] + bv;
        if (RELU) x = x > 0.f ? x : 0.f;
        out[(long long)gr * HIDN + col] = f2bf(x);
      }
    }
  }
}

// ---------------- fused edge classifier ----------------
// out[e] = relu(concat(zU[row],zM[col]) @ Wc1^T + bc1) . Wc2 + bc2
// Block: 128 threads, 16 edges, K=256, Wc2 dot fused into epilogue.
__global__ __launch_bounds__(128)
void k_edge_cls(const int* __restrict__ tgt, int Et,
                const unsigned short* __restrict__ zU, const unsigned short* __restrict__ zM,
                const unsigned short* __restrict__ wc1, const float* __restrict__ bc1,
                const float* __restrict__ wc2, const float* __restrict__ bc2,
                float* __restrict__ out, int nU, int nM) {
  constexpr int K = 256;
  __shared__ __align__(16) unsigned short At[16 * K];
  __shared__ int uid[16];
  __shared__ int mid[16];
  __shared__ float rowsum[16];
  const int tid = threadIdx.x;
  const int eBase = blockIdx.x * 16;

  if (tid < 16) {
    int e = eBase + tid;
    int u = 0, m = 0;
    if (e < Et) { u = tgt[e]; m = tgt[Et + e]; }
    uid[tid] = ((unsigned)u < (unsigned)nU) ? u : 0;
    mid[tid] = ((unsigned)m < (unsigned)nM) ? m : 0;
    rowsum[tid] = 0.f;
  }
  __syncthreads();

  // gather 16 rows x (128|128) bf16 via 128-bit chunks
  for (int i = tid; i < 16 * 32; i += 128) {
    int r = i >> 5, seg = i & 31;
    const unsigned short* src = (seg < 16)
        ? &zU[(long long)uid[r] * HIDN + seg * 8]
        : &zM[(long long)mid[r] * HIDN + (seg - 16) * 8];
    *(u32x4*)&At[r * K + seg * 8] = *(const u32x4*)src;
  }
  __syncthreads();

  const int lane = tid & 31, wave = tid >> 5;
  const int lrow = lane & 15, hi = lane >> 4;
  v8f acc0 = {0,0,0,0,0,0,0,0};
  v8f acc1 = {0,0,0,0,0,0,0,0};

  for (int k0 = 0; k0 < K; k0 += 32) {
    FragBF a;
    const unsigned short* ar = &At[lrow * K + k0 + hi * 16];
    a.q[0] = *(const u32x4*)ar;
    a.q[1] = *(const u32x4*)(ar + 8);
    {
      int n = (wave * 2 + 0) * 16 + lrow;
      const unsigned short* br = &wc1[(long long)n * K + k0 + hi * 16];
      FragBF b; b.q[0] = *(const u32x4*)br; b.q[1] = *(const u32x4*)(br + 8);
      acc0 = __builtin_amdgcn_wmma_f32_16x16x32_bf16(false, a.v, false, b.v,
                                                     (short)0, acc0, false, false);
    }
    {
      int n = (wave * 2 + 1) * 16 + lrow;
      const unsigned short* br = &wc1[(long long)n * K + k0 + hi * 16];
      FragBF b; b.q[0] = *(const u32x4*)br; b.q[1] = *(const u32x4*)(br + 8);
      acc1 = __builtin_amdgcn_wmma_f32_16x16x32_bf16(false, a.v, false, b.v,
                                                     (short)0, acc1, false, false);
    }
  }

  // epilogue: relu(+bc1) * wc2 -> per-row reduction in LDS
  #pragma unroll
  for (int t = 0; t < 2; ++t) {
    v8f acc = t ? acc1 : acc0;
    int col = (wave * 2 + t) * 16 + lrow;
    float bv = bc1[col], wv = wc2[col];
    #pragma unroll
    for (int v = 0; v < 8; ++v) {
      float x = acc[v] + bv;
      x = x > 0.f ? x : 0.f;
      int r = hi ? (v + 8) : v;
      atomicAdd(&rowsum[r], x * wv);
    }
  }
  __syncthreads();
  if (tid < 16 && (eBase + tid) < Et) out[eBase + tid] = rowsum[tid] + bc2[0];
}

// ---------------- host launch ----------------

extern "C" void kernel_launch(void* const* d_in, const int* in_sizes, int n_in,
                              void* d_out, int out_size, void* d_ws, size_t ws_size,
                              hipStream_t stream) {
  const float* x_user  = (const float*)d_in[0];
  const float* x_merch = (const float*)d_in[1];
  const int* e_um = (const int*)d_in[2];
  const int* e_mu = (const int*)d_in[3];
  const int* tgt  = (const int*)d_in[4];
  const float* W1_um_l = (const float*)d_in[5];  const float* b1_um_l = (const float*)d_in[6];
  const float* W1_um_r = (const float*)d_in[7];
  const float* W1_mu_l = (const float*)d_in[8];  const float* b1_mu_l = (const float*)d_in[9];
  const float* W1_mu_r = (const float*)d_in[10];
  const float* W2_um_l = (const float*)d_in[11]; const float* b2_um_l = (const float*)d_in[12];
  const float* W2_um_r = (const float*)d_in[13];
  const float* W2_mu_l = (const float*)d_in[14]; const float* b2_mu_l = (const float*)d_in[15];
  const float* W2_mu_r = (const float*)d_in[16];
  const float* Wc1 = (const float*)d_in[17]; const float* bc1 = (const float*)d_in[18];
  const float* Wc2 = (const float*)d_in[19]; const float* bc2 = (const float*)d_in[20];
  float* out = (float*)d_out;

  const int Eum = in_sizes[2] / 2;
  const int Emu = in_sizes[3] / 2;
  const int Et  = in_sizes[4] / 2;

  // ---- workspace layout (all 16B aligned) ----
  float* aggM = (float*)d_ws;                              // 50000*128 f32
  float* aggU = aggM + (size_t)NMERCH * HIDN;              // 50000*128 f32
  float* cntM = aggU + (size_t)NUSER * HIDN;               // 50000 f32
  float* cntU = cntM + NMERCH;                             // 50000 f32
  unsigned short* hU = (unsigned short*)(cntU + NUSER);    // 50000*128 bf16
  unsigned short* hM  = hU + (size_t)NUSER * HIDN;
  unsigned short* zU  = hM + (size_t)NMERCH * HIDN;
  unsigned short* zM  = zU + (size_t)NUSER * HIDN;
  unsigned short* wb  = zM + (size_t)NMERCH * HIDN;        // bf16 weights
  unsigned short* w1umlB = wb;                 // 128*64
  unsigned short* w1umrB = w1umlB + 128 * 64;
  unsigned short* w1mulB = w1umrB + 128 * 64;
  unsigned short* w1murB = w1mulB + 128 * 64;
  unsigned short* w2umlB = w1murB + 128 * 64;  // 128*128
  unsigned short* w2umrB = w2umlB + 128 * 128;
  unsigned short* w2mulB = w2umrB + 128 * 128;
  unsigned short* w2murB = w2mulB + 128 * 128;
  unsigned short* wc1B   = w2murB + 128 * 128; // 128*256

  // 1) zero agg + cnt
  long long nZ1 = (long long)(NMERCH + NUSER) * HIDN + NMERCH + NUSER;
  k_zero_f32<<<4096, 256, 0, stream>>>(aggM, nZ1);

  // 2) weights -> bf16
  k_cvt_bf16<<<(8192 + 255) / 256, 256, 0, stream>>>(W1_um_l, w1umlB, 8192);
  k_cvt_bf16<<<(8192 + 255) / 256, 256, 0, stream>>>(W1_um_r, w1umrB, 8192);
  k_cvt_bf16<<<(8192 + 255) / 256, 256, 0, stream>>>(W1_mu_l, w1mulB, 8192);
  k_cvt_bf16<<<(8192 + 255) / 256, 256, 0, stream>>>(W1_mu_r, w1murB, 8192);
  k_cvt_bf16<<<(16384 + 255) / 256, 256, 0, stream>>>(W2_um_l, w2umlB, 16384);
  k_cvt_bf16<<<(16384 + 255) / 256, 256, 0, stream>>>(W2_um_r, w2umrB, 16384);
  k_cvt_bf16<<<(16384 + 255) / 256, 256, 0, stream>>>(W2_mu_l, w2mulB, 16384);
  k_cvt_bf16<<<(16384 + 255) / 256, 256, 0, stream>>>(W2_mu_r, w2murB, 16384);
  k_cvt_bf16<<<(32768 + 255) / 256, 256, 0, stream>>>(Wc1, wc1B, 32768);

  // 3) edge counts (reused by both layers: same edge index)
  k_count<<<(Eum + 255) / 256, 256, 0, stream>>>(e_um, Eum, cntM, NMERCH);
  k_count<<<(Emu + 255) / 256, 256, 0, stream>>>(e_mu, Emu, cntU, NUSER);

  // 4) layer-1 scatter (f32 features, D=64)
  {
    long long t = (long long)Eum * DIN;
    k_scatter<DIN, true><<<(unsigned)((t + 255) / 256), 256, 0, stream>>>(
        e_um, Eum, x_user, aggM, NUSER, NMERCH);
    t = (long long)Emu * DIN;
    k_scatter<DIN, true><<<(unsigned)((t + 255) / 256), 256, 0, stream>>>(
        e_mu, Emu, x_merch, aggU, NMERCH, NUSER);
  }

  // 5) layer-1 fused SAGE GEMMs (relu)
  k_sage_gemm<DIN, true, true><<<(NMERCH + 15) / 16, 128, 0, stream>>>(
      aggM, cntM, x_merch, w1umlB, w1umrB, b1_um_l, hM, NMERCH);
  k_sage_gemm<DIN, true, true><<<(NUSER + 15) / 16, 128, 0, stream>>>(
      aggU, cntU, x_user, w1mulB, w1murB, b1_mu_l, hU, NUSER);

  // 6) re-zero agg (counts preserved)
  k_zero_f32<<<4096, 256, 0, stream>>>(aggM, (long long)(NMERCH + NUSER) * HIDN);

  // 7) layer-2 scatter (bf16 hidden, D=128)
  {
    long long t = (long long)Eum * HIDN;
    k_scatter<HIDN, false><<<(unsigned)((t + 255) / 256), 256, 0, stream>>>(
        e_um, Eum, hU, aggM, NUSER, NMERCH);
    t = (long long)Emu * HIDN;
    k_scatter<HIDN, false><<<(unsigned)((t + 255) / 256), 256, 0, stream>>>(
        e_mu, Emu, hM, aggU, NMERCH, NUSER);
  }

  // 8) layer-2 fused SAGE GEMMs (no relu)
  k_sage_gemm<HIDN, false, false><<<(NMERCH + 15) / 16, 128, 0, stream>>>(
      aggM, cntM, hM, w2umlB, w2umrB, b2_um_l, zM, NMERCH);
  k_sage_gemm<HIDN, false, false><<<(NUSER + 15) / 16, 128, 0, stream>>>(
      aggU, cntU, hU, w2mulB, w2murB, b2_mu_l, zU, NUSER);

  // 9) fused edge classifier (gather + GEMM + relu + Wc2 dot + bc2)
  k_edge_cls<<<(Et + 15) / 16, 128, 0, stream>>>(
      tgt, Et, zU, zM, wc1B, bc1, Wc2, bc2, out, NUSER, NMERCH);
}
